// RNN_17214228922840
// MI455X (gfx1250) — compile-verified, matrix operand order checked by reading, not run on hardware
//
#include <hip/hip_runtime.h>
#include <hip/hip_bf16.h>

typedef __attribute__((ext_vector_type(16))) _Float16 v16h;
typedef __attribute__((ext_vector_type(8)))  _Float16 v8h;
typedef __attribute__((ext_vector_type(8)))  float    v8f;

#define BB    128
#define TV    32
#define TE    64
#define VOCAB 1024
#define EMB   256
#define HH    512
#define GG    (3*HH)
#define RR    128
#define NET   8

// 16x32 f16 fragment from row-major [rows, ld]; matches ISA 7.12.2 A layout,
// and B layout when W is stored [N,K] row-major.
__device__ __forceinline__ v16h load_frag(const _Float16* __restrict__ tile, int ld) {
  const int lane = threadIdx.x & 31;
  const int m  = lane & 15;
  const int ko = (lane >> 4) << 3;
  const _Float16* r = tile + (size_t)m * ld + ko;
  v8h lo = *reinterpret_cast<const v8h*>(r);
  v8h hi = *reinterpret_cast<const v8h*>(r + 16);
  v16h f;
#pragma unroll
  for (int i = 0; i < 8; ++i) { f[i] = lo[i]; f[i + 8] = hi[i]; }
  return f;
}

__device__ __forceinline__ v8f wmma32(v16h a, v16h b, v8f c) {
  return __builtin_amdgcn_wmma_f32_16x16x32_f16(false, a, false, b,
                                                (short)0, c, false, false);
}

enum { OUT_F16 = 0, OUT_F32_BIAS = 1, OUT_BVT = 2 };

// Register-blocked GEMM: each wave computes a 32(M) x 64(N) block.
// C = A[M,K] * W[N,K]^T (+bias). M%32==0, N%64==0, K%32==0.
template <int MODE>
__global__ void gemm_wmma_blk(const _Float16* __restrict__ A,
                              const _Float16* __restrict__ W,
                              const float* __restrict__ bias,
                              void* __restrict__ Cout,
                              int M, int N, int K, int T, int Bdim) {
  const int tilesN = N >> 6;
  const int tiles  = (M >> 5) * tilesN;
  const int w = blockIdx.x * (blockDim.x >> 5) + (threadIdx.x >> 5);
  if (w >= tiles) return;
  const int tm = w / tilesN, tn = w % tilesN;
  const _Float16* A0 = A + (size_t)(tm * 32) * K;
  const _Float16* A1 = A0 + (size_t)16 * K;
  const _Float16* W0 = W + (size_t)(tn * 64) * K;

  v8f acc[2][4] = {};
  for (int k0 = 0; k0 < K; k0 += 32) {
    __builtin_prefetch(A0 + k0 + 256, 0, 3);
    __builtin_prefetch(W0 + k0 + 256, 0, 3);
    v16h a0 = load_frag(A0 + k0, K);
    v16h a1 = load_frag(A1 + k0, K);
#pragma unroll
    for (int j = 0; j < 4; ++j) {
      v16h b = load_frag(W0 + (size_t)(j * 16) * K + k0, K);
      acc[0][j] = wmma32(a0, b, acc[0][j]);
      acc[1][j] = wmma32(a1, b, acc[1][j]);
    }
  }

  const int lane = threadIdx.x & 31;
  const int n  = lane & 15;
  const int mo = (lane >> 4) << 3;
#pragma unroll
  for (int i = 0; i < 2; ++i) {
#pragma unroll
    for (int j = 0; j < 4; ++j) {
#pragma unroll
      for (int v = 0; v < 8; ++v) {
        int row = tm * 32 + i * 16 + mo + v;
        int col = tn * 64 + j * 16 + n;
        if (MODE == OUT_F16) {
          ((_Float16*)Cout)[(size_t)row * N + col] = (_Float16)acc[i][j][v];
        } else if (MODE == OUT_F32_BIAS) {
          ((float*)Cout)[(size_t)row * N + col] = acc[i][j][v] + bias[col];
        } else { // OUT_BVT: row = t*B+b, store [b, col, t]
          int t = row / Bdim, b = row % Bdim;
          ((float*)Cout)[((size_t)b * N + col) * T + t] = acc[i][j][v] + bias[col];
        }
      }
    }
  }
}

// Fused GRU step: gh = h*Wh^T for 3 gates + gate nonlinearity.
// Each wave: 32 batch rows x 16 h columns, 3 gates (6 WMMA per K-step).
__global__ void gru_step_wmma(const _Float16* __restrict__ hprev,  // [B,H]
                              const _Float16* __restrict__ Wh,     // [3H,H]
                              const _Float16* __restrict__ gi,     // [B,3H]
                              const float* __restrict__ bi,
                              const float* __restrict__ bh,
                              _Float16* __restrict__ hout) {       // [B,H]
  const int tilesN = HH >> 4;                 // 32
  const int tiles  = (BB >> 5) * tilesN;      // 4 * 32 = 128 waves
  const int w = blockIdx.x * (blockDim.x >> 5) + (threadIdx.x >> 5);
  if (w >= tiles) return;
  const int tm = w / tilesN, tn = w % tilesN;
  const _Float16* A0 = hprev + (size_t)(tm * 32) * HH;
  const _Float16* A1 = A0 + (size_t)16 * HH;

  v8f acc[2][3] = {};
  for (int k0 = 0; k0 < HH; k0 += 32) {
    v16h a0 = load_frag(A0 + k0, HH);
    v16h a1 = load_frag(A1 + k0, HH);
#pragma unroll
    for (int g = 0; g < 3; ++g) {
      v16h b = load_frag(Wh + ((size_t)(g * HH + tn * 16)) * HH + k0, HH);
      acc[0][g] = wmma32(a0, b, acc[0][g]);
      acc[1][g] = wmma32(a1, b, acc[1][g]);
    }
  }

  const int lane = threadIdx.x & 31;
  const int n  = lane & 15;
  const int mo = (lane >> 4) << 3;
#pragma unroll
  for (int i = 0; i < 2; ++i) {
#pragma unroll
    for (int v = 0; v < 8; ++v) {
      int b = tm * 32 + i * 16 + mo + v;
      int h = tn * 16 + n;
      float ir = (float)gi[(size_t)b * GG + h]          + bi[h];
      float iz = (float)gi[(size_t)b * GG + HH + h]     + bi[HH + h];
      float in = (float)gi[(size_t)b * GG + 2 * HH + h] + bi[2 * HH + h];
      float hr = acc[i][0][v] + bh[h];
      float hz = acc[i][1][v] + bh[HH + h];
      float hn = acc[i][2][v] + bh[2 * HH + h];
      float r  = 1.f / (1.f + __expf(-(ir + hr)));
      float z  = 1.f / (1.f + __expf(-(iz + hz)));
      float nn = tanhf(in + r * hn);
      float hp = (float)hprev[(size_t)b * HH + h];
      hout[(size_t)b * HH + h] = (_Float16)((1.f - z) * nn + z * hp);
    }
  }
}

__global__ void cvt_f32_f16(const float* __restrict__ s, _Float16* __restrict__ d, int n) {
  int i = blockIdx.x * blockDim.x + threadIdx.x;
  if (i < n) d[i] = (_Float16)s[i];
}

__global__ void zero_f16(_Float16* __restrict__ p, int n) {
  int i = blockIdx.x * blockDim.x + threadIdx.x;
  if (i < n) p[i] = (_Float16)0.f;
}

__global__ void gather_vertex(const int* __restrict__ iv, const float* __restrict__ emb,
                              _Float16* __restrict__ X) {
  int idx = blockIdx.x * blockDim.x + threadIdx.x;
  if (idx >= TV * BB * EMB) return;
  int k = idx % EMB, row = idx / EMB;
  int b = row % BB, t = row / BB;
  int tok = iv[b * TV + t];
  X[idx] = (_Float16)emb[(size_t)tok * EMB + k];
}

__global__ void gather_edge(const int* __restrict__ ie, const float* __restrict__ emb,
                            _Float16* __restrict__ X) {
  int idx = blockIdx.x * blockDim.x + threadIdx.x;
  if (idx >= TE * BB * 2 * EMB) return;
  int k = idx % (2 * EMB), row = idx / (2 * EMB);
  int b = row % BB, t = row / BB;
  int which = k < EMB ? 0 : 1;
  int tok = ie[((size_t)b * TE + t) * 3 + which];
  X[idx] = (_Float16)emb[(size_t)tok * EMB + (k & (EMB - 1))];
}

__global__ void etype_proj(const float* __restrict__ E,     // [TE*B, RR]
                           const float* __restrict__ W,     // [NET, RR]
                           const float* __restrict__ bias,  // [NET]
                           float* __restrict__ out) {       // [B, NET, TE]
  int idx = blockIdx.x * blockDim.x + threadIdx.x;
  if (idx >= BB * NET * TE) return;
  int t = idx % TE, k = (idx / TE) % NET, b = idx / (TE * NET);
  const float* er = E + ((size_t)t * BB + b) * RR;
  const float* wr = W + (size_t)k * RR;
  float s = bias[k];
#pragma unroll 4
  for (int r = 0; r < RR; ++r) s += er[r] * wr[r];
  out[idx] = s;
}

__global__ void attn_edge(const float* __restrict__ E,      // [TE*B, RR]
                          const float* __restrict__ V,      // [TV*B, RR]
                          const float* __restrict__ sw, const float* __restrict__ sb,
                          const float* __restrict__ dw, const float* __restrict__ db,
                          float* __restrict__ out) {        // [B, TV, 2, TE]
  __shared__ float vr[RR], swv[RR], dwv[RR];
  int b  = blockIdx.x / TV;
  int tv = blockIdx.x % TV;
  for (int i = threadIdx.x; i < RR; i += blockDim.x) {
    vr[i]  = V[((size_t)tv * BB + b) * RR + i];
    swv[i] = sw[i];
    dwv[i] = dw[i];
  }
  __syncthreads();
  int te = threadIdx.x;
  const float* er = E + ((size_t)te * BB + b) * RR;
  float ss = 0.f, sd = 0.f;
#pragma unroll 4
  for (int r = 0; r < RR; ++r) {
    float u = tanhf(er[r] + vr[r]);
    ss += u * swv[r];
    sd += u * dwv[r];
  }
  size_t base = (((size_t)b * TV + tv) * 2) * TE;
  out[base + te]      = ss + sb[0];
  out[base + TE + te] = sd + db[0];
}

extern "C" void kernel_launch(void* const* d_in, const int* in_sizes, int n_in,
                              void* d_out, int out_size, void* d_ws, size_t ws_size,
                              hipStream_t stream) {
  const int*   iv      = (const int*)  d_in[0];
  const int*   ie      = (const int*)  d_in[1];
  const float* vemb    = (const float*)d_in[2];
  const float* v_Wi    = (const float*)d_in[3];
  const float* v_Wh    = (const float*)d_in[4];
  const float* v_bi    = (const float*)d_in[5];
  const float* v_bh    = (const float*)d_in[6];
  const float* vout_W  = (const float*)d_in[7];
  const float* vout_b  = (const float*)d_in[8];
  const float* vrep_W  = (const float*)d_in[9];
  const float* vrep_b  = (const float*)d_in[10];
  const float* eemb    = (const float*)d_in[11];
  const float* e_Wi    = (const float*)d_in[12];
  const float* e_Wh    = (const float*)d_in[13];
  const float* e_bi    = (const float*)d_in[14];
  const float* e_bh    = (const float*)d_in[15];
  const float* erep_W  = (const float*)d_in[16];
  const float* erep_b  = (const float*)d_in[17];
  const float* etype_W = (const float*)d_in[18];
  const float* etype_b = (const float*)d_in[19];
  const float* src_w   = (const float*)d_in[20];
  const float* src_b   = (const float*)d_in[21];
  const float* dst_w   = (const float*)d_in[22];
  const float* dst_b   = (const float*)d_in[23];

  float* out = (float*)d_out;
  float* out_vertex = out;
  float* out_edge   = out + (size_t)BB * VOCAB * TV;
  float* out_etype  = out_edge + (size_t)BB * TV * 2 * TE;

  char* ws = (char*)d_ws;
  size_t off = 0;
  auto carve = [&](size_t bytes) -> void* {
    void* p = ws + off;
    off += (bytes + 255) & ~(size_t)255;
    return p;
  };
  _Float16* vWi16   = (_Float16*)carve((size_t)GG * EMB * 2);
  _Float16* vWh16   = (_Float16*)carve((size_t)GG * HH * 2);
  _Float16* eWi16   = (_Float16*)carve((size_t)GG * 2 * EMB * 2);
  _Float16* eWh16   = (_Float16*)carve((size_t)GG * HH * 2);
  _Float16* voutW16 = (_Float16*)carve((size_t)VOCAB * HH * 2);
  _Float16* vrepW16 = (_Float16*)carve((size_t)RR * HH * 2);
  _Float16* erepW16 = (_Float16*)carve((size_t)RR * HH * 2);
  _Float16* Xv16    = (_Float16*)carve((size_t)TV * BB * EMB * 2);
  _Float16* Xe16    = (_Float16*)carve((size_t)TE * BB * 2 * EMB * 2);
  _Float16* giV     = (_Float16*)carve((size_t)TV * BB * GG * 2);
  _Float16* giE     = (_Float16*)carve((size_t)TE * BB * GG * 2);
  _Float16* hsV     = (_Float16*)carve((size_t)(TV + 1) * BB * HH * 2);
  _Float16* hsE     = (_Float16*)carve((size_t)(TE + 1) * BB * HH * 2);
  float*    vrepF   = (float*)   carve((size_t)TV * BB * RR * 4);
  float*    eF      = (float*)   carve((size_t)TE * BB * RR * 4);

  auto cvt = [&](const float* s, _Float16* d, int n) {
    cvt_f32_f16<<<(n + 255) / 256, 256, 0, stream>>>(s, d, n);
  };

  cvt(v_Wi,   vWi16,   GG * EMB);
  cvt(v_Wh,   vWh16,   GG * HH);
  cvt(e_Wi,   eWi16,   GG * 2 * EMB);
  cvt(e_Wh,   eWh16,   GG * HH);
  cvt(vout_W, voutW16, VOCAB * HH);
  cvt(vrep_W, vrepW16, RR * HH);
  cvt(erep_W, erepW16, RR * HH);

  gather_vertex<<<(TV * BB * EMB + 255) / 256, 256, 0, stream>>>(iv, vemb, Xv16);
  gather_edge<<<(TE * BB * 2 * EMB + 255) / 256, 256, 0, stream>>>(ie, eemb, Xe16);

  zero_f16<<<(BB * HH + 255) / 256, 256, 0, stream>>>(hsV, BB * HH);
  zero_f16<<<(BB * HH + 255) / 256, 256, 0, stream>>>(hsE, BB * HH);

  // input projections gi = X * Wi^T
  {
    int tiles = (TV * BB / 32) * (GG / 64);   // 128*24 = 3072
    gemm_wmma_blk<OUT_F16><<<(tiles + 3) / 4, 128, 0, stream>>>(
        Xv16, vWi16, nullptr, giV, TV * BB, GG, EMB, 0, 0);
  }
  {
    int tiles = (TE * BB / 32) * (GG / 64);   // 256*24 = 6144
    gemm_wmma_blk<OUT_F16><<<(tiles + 3) / 4, 128, 0, stream>>>(
        Xe16, eWi16, nullptr, giE, TE * BB, GG, 2 * EMB, 0, 0);
  }

  // sequential GRU recurrences (fused WMMA + gate math)
  for (int t = 0; t < TV; ++t) {
    gru_step_wmma<<<32, 128, 0, stream>>>(hsV + (size_t)t * BB * HH, vWh16,
                                          giV + (size_t)t * BB * GG, v_bi, v_bh,
                                          hsV + (size_t)(t + 1) * BB * HH);
  }
  for (int t = 0; t < TE; ++t) {
    gru_step_wmma<<<32, 128, 0, stream>>>(hsE + (size_t)t * BB * HH, eWh16,
                                          giE + (size_t)t * BB * GG, e_bi, e_bh,
                                          hsE + (size_t)(t + 1) * BB * HH);
  }

  // O_vertex = hv * vout_W^T + b  -> [b, v, t]
  {
    int tiles = (TV * BB / 32) * (VOCAB / 64);  // 128*16 = 2048
    gemm_wmma_blk<OUT_BVT><<<(tiles + 3) / 4, 128, 0, stream>>>(
        hsV + (size_t)BB * HH, voutW16, vout_b, out_vertex,
        TV * BB, VOCAB, HH, TV, BB);
  }
  // vrep / e projections
  {
    int tiles = (TV * BB / 32) * (RR / 64);     // 128*2 = 256
    gemm_wmma_blk<OUT_F32_BIAS><<<(tiles + 3) / 4, 128, 0, stream>>>(
        hsV + (size_t)BB * HH, vrepW16, vrep_b, vrepF, TV * BB, RR, HH, 0, 0);
  }
  {
    int tiles = (TE * BB / 32) * (RR / 64);     // 256*2 = 512
    gemm_wmma_blk<OUT_F32_BIAS><<<(tiles + 3) / 4, 128, 0, stream>>>(
        hsE + (size_t)BB * HH, erepW16, erep_b, eF, TE * BB, RR, HH, 0, 0);
  }

  etype_proj<<<(BB * NET * TE + 255) / 256, 256, 0, stream>>>(eF, etype_W, etype_b, out_etype);

  attn_edge<<<BB * TV, TE, 0, stream>>>(eF, vrepF, src_w, src_b, dst_w, dst_b, out_edge);
}
